// QuantAttention_2637109920151
// MI455X (gfx1250) — matrix operand order repaired and checked by reading, not compiled, not run on hardware
//
#include <hip/hip_runtime.h>
#include <hip/hip_bf16.h>

// ---------------------------------------------------------------------------
// Window attention (B=256, N=196, C=512, H=16, hd=32) for gfx1250 / MI455X.
// bf16 WMMA for all GEMMs (fp32 accumulate), fused softmax attention, and
// CDNA5 async global->LDS staging (ASYNCcnt) for the projection GEMMs.
// ---------------------------------------------------------------------------

typedef __bf16 bf16;
typedef __attribute__((ext_vector_type(16))) __bf16 v16bf;
typedef __attribute__((ext_vector_type(8)))  float  v8f;
typedef __attribute__((ext_vector_type(4)))  int    v4i;

#define WMMA_BF16(a, b, c) \
  __builtin_amdgcn_wmma_f32_16x16x32_bf16(false, (a), false, (b), (short)0, (c), false, false)

static constexpr int BATCH  = 256;
static constexpr int NTOK   = 196;
static constexpr int CDIM   = 512;
static constexpr int HEADS  = 16;
static constexpr int HD     = 32;
static constexpr int NPADQ  = 208;   // query-row padding (multiple of 16)
static constexpr int NPADK  = 224;   // key/col padding (multiple of 32)
static constexpr int MROWS  = BATCH * NTOK;        // 50176, multiple of 128
static constexpr int MTILES = MROWS / 16;          // 3136

// ---- workspace layout (bytes; every size is a multiple of 256) -------------
static constexpr size_t XB_BYTES   = (size_t)MROWS * CDIM * 2;               // also xo
static constexpr size_t WQB_BYTES  = (size_t)3 * CDIM * CDIM * 2;
static constexpr size_t PWB_BYTES  = (size_t)CDIM * CDIM * 2;
static constexpr size_t QS_BYTES   = (size_t)BATCH * HEADS * NPADQ * HD * 2;
static constexpr size_t KS_BYTES   = (size_t)BATCH * HEADS * NPADK * HD * 2;
static constexpr size_t VT_BYTES   = (size_t)BATCH * HEADS * HD * NPADK * 2;
static constexpr size_t BIAS_BYTES = (size_t)HEADS * NPADQ * NPADK * 4;

static constexpr size_t O_XB   = 0;
static constexpr size_t O_WQB  = O_XB  + XB_BYTES;
static constexpr size_t O_PWB  = O_WQB + WQB_BYTES;
static constexpr size_t O_QS   = O_PWB + PWB_BYTES;
static constexpr size_t O_KS   = O_QS  + QS_BYTES;
static constexpr size_t O_VT   = O_KS  + KS_BYTES;
static constexpr size_t O_BIAS = O_VT  + VT_BYTES;

// ---------------------------------------------------------------------------
// CDNA5 async global->LDS copy (tracked by ASYNCcnt) + wait.
// Builtin signature (from hipcc diagnostic): param0 = global int4*, so we cast
// through integer round-trips: 64-bit for AS1, low-32 LDS offset for AS3.
// ---------------------------------------------------------------------------
__device__ __forceinline__ void async_b128(const bf16* g, bf16* l) {
#if __has_builtin(__builtin_amdgcn_global_load_async_to_lds_b128)
  __builtin_amdgcn_global_load_async_to_lds_b128(
      (__attribute__((address_space(1))) v4i*)(unsigned long long)(g),
      (__attribute__((address_space(3))) v4i*)(unsigned)(unsigned long long)(l),
      0, 0);
#else
  asm volatile("global_load_async_to_lds_b128 %0, %1, off"
               :: "v"((unsigned)(unsigned long long)l),
                  "v"((unsigned long long)g)
               : "memory");
#endif
}

__device__ __forceinline__ void wait_async0() {
#if __has_builtin(__builtin_amdgcn_s_wait_asynccnt)
  __builtin_amdgcn_s_wait_asynccnt(0);
#else
  asm volatile("s_wait_asynccnt 0x0" ::: "memory");
#endif
}

// ---------------------------------------------------------------------------
// Operand loaders (wave32), contiguous k order k = 16*half + s, same for A and B.
// A 16x32 (16-bit): lane<16 row r holds K 0-15; lane>=16 holds K 16-31
// B 32x16 (16-bit): lane<16 col n holds K 0-15; lane>=16 holds K 16-31
// ---------------------------------------------------------------------------
__device__ __forceinline__ v16bf load_a_tile(const bf16* base, int stride, int lane) {
  const int r = lane & 15;
  const int half = lane >> 4;
  const bf16* row = base + (size_t)r * stride;
  union { uint4 u[2]; v16bf v; } t;
  t.u[0] = *(const uint4*)(row + half * 16);
  t.u[1] = *(const uint4*)(row + half * 16 + 8);
  return t.v;
}

__device__ __forceinline__ v16bf load_b_tile(const bf16* base, int stride, int lane) {
  const int n = lane & 15;
  const int half = lane >> 4;
  const bf16* row = base + (size_t)n * stride + half * 16;
  union { uint4 u[2]; v16bf v; } t;
  t.u[0] = *(const uint4*)(row + 0);
  t.u[1] = *(const uint4*)(row + 8);
  return t.v;
}

// Stage a 128-row x 32-col bf16 tile (rows of `gbase` w/ stride, cols kk..kk+31)
// into LDS (row-major, stride 32). 256 threads -> 2 async B128 copies each.
__device__ __forceinline__ void stage_tile(const bf16* gbase, int gstride, int kk,
                                           bf16* lds, int tid) {
#pragma unroll
  for (int q = 0; q < 2; q++) {
    int c = tid * 2 + q;            // 0..511 (16B chunks)
    int row = c >> 2;               // 0..127
    int off = (c & 3) * 8;          // element offset within 32-col row
    async_b128(gbase + (size_t)row * gstride + kk + off, lds + row * 32 + off);
  }
}

// ---------------------------------------------------------------------------
__global__ void cast_f32_to_bf16(const float* __restrict__ src, bf16* __restrict__ dst, int n) {
  int i = blockIdx.x * blockDim.x + threadIdx.x;
  if (i < n) dst[i] = (bf16)src[i];
}

__global__ void zero_u4(uint4* __restrict__ p, long n4) {
  long i = (long)blockIdx.x * blockDim.x + threadIdx.x;
  long stride = (long)gridDim.x * blockDim.x;
  uint4 z = {0u, 0u, 0u, 0u};
  for (; i < n4; i += stride) p[i] = z;
}

// bias_pad[h][m][n] = (m<196 && n<196) ? table[idx[m][n]][h] : -1e30
__global__ void build_bias(const float* __restrict__ table, const int* __restrict__ idx,
                           float* __restrict__ bias) {
  int t = blockIdx.x * blockDim.x + threadIdx.x;
  const int total = HEADS * NPADQ * NPADK;
  if (t >= total) return;
  int n = t % NPADK;
  int m = (t / NPADK) % NPADQ;
  int h = t / (NPADK * NPADQ);
  float v = -1.0e30f;
  if (m < NTOK && n < NTOK) v = table[idx[m * NTOK + n] * HEADS + h];
  bias[t] = v;
}

// ---------------------------------------------------------------------------
// QKV GEMM: workgroup computes a 128x128 tile of [50176,512]x[1536,512]^T.
// Double-buffered async global->LDS staging; 8 waves x (32x64 subtile).
// Epilogue scatters q (scaled, padded), k (padded), v transposed (padded).
// ---------------------------------------------------------------------------
__global__ void __launch_bounds__(256) qkv_gemm(const bf16* __restrict__ xb,
                                                const bf16* __restrict__ wqb,
                                                bf16* __restrict__ qs,
                                                bf16* __restrict__ ksb,
                                                bf16* __restrict__ vt) {
  __shared__ __align__(16) bf16 ldsA[2][128 * 32];
  __shared__ __align__(16) bf16 ldsB[2][128 * 32];
  const int tid = threadIdx.x;
  const int lane = tid & 31;
  const int wid = tid >> 5;
  const int wm = wid & 3;          // 4 M-groups of 32 rows
  const int wn = wid >> 2;         // 2 N-groups of 64 cols
  const int m0 = blockIdx.x * 128;
  const int n0 = blockIdx.y * 128;

  v8f acc[2][4] = {};

  stage_tile(xb + (size_t)m0 * CDIM, CDIM, 0, ldsA[0], tid);
  stage_tile(wqb + (size_t)n0 * CDIM, CDIM, 0, ldsB[0], tid);
  wait_async0();
  __syncthreads();

  const int KSTEPS = CDIM / 32;
  for (int step = 0; step < KSTEPS; step++) {
    const int cur = step & 1;
    if (step + 1 < KSTEPS) {
      const int kk = (step + 1) * 32;
      stage_tile(xb + (size_t)m0 * CDIM, CDIM, kk, ldsA[cur ^ 1], tid);
      stage_tile(wqb + (size_t)n0 * CDIM, CDIM, kk, ldsB[cur ^ 1], tid);
    }
    v16bf a0 = load_a_tile(ldsA[cur] + (wm * 32 + 0) * 32, 32, lane);
    v16bf a1 = load_a_tile(ldsA[cur] + (wm * 32 + 16) * 32, 32, lane);
#pragma unroll
    for (int t = 0; t < 4; t++) {
      v16bf bt = load_b_tile(ldsB[cur] + (wn * 64 + t * 16) * 32, 32, lane);
      acc[0][t] = WMMA_BF16(a0, bt, acc[0][t]);
      acc[1][t] = WMMA_BF16(a1, bt, acc[1][t]);
    }
    wait_async0();
    __syncthreads();
  }

  const int half = lane >> 4;
  const int nl = lane & 15;
  const float scale = 0.17677669529663687f;  // 32^-0.5
#pragma unroll
  for (int a = 0; a < 2; a++) {
#pragma unroll
    for (int t = 0; t < 4; t++) {
#pragma unroll
      for (int r = 0; r < 8; r++) {
        int m = m0 + wm * 32 + a * 16 + r + 8 * half;   // < 50176 always
        int j = n0 + wn * 64 + t * 16 + nl;             // < 1536
        int bidx = m / NTOK, tok = m % NTOK;
        int sel = j >> 9, rem = j & 511;
        int h = rem >> 5, d = rem & 31;
        size_t bh = (size_t)bidx * HEADS + h;
        float val = acc[a][t][r];
        if (sel == 0)      qs[(bh * NPADQ + tok) * HD + d]  = (bf16)(val * scale);
        else if (sel == 1) ksb[(bh * NPADK + tok) * HD + d] = (bf16)val;
        else               vt[(bh * HD + d) * NPADK + tok]  = (bf16)val;
      }
    }
  }
}

// ---------------------------------------------------------------------------
// Fused attention: one wave per (query row-block i, head h, batch b).
// S = q*k^T (14 WMMAs) + bias, softmax in registers (16-lane shfl reductions),
// P -> LDS bf16, O = P*V (14 WMMAs), deferred 1/rowsum normalization.
// ---------------------------------------------------------------------------
__global__ void __launch_bounds__(32) attn_fused(const bf16* __restrict__ qs,
                                                 const bf16* __restrict__ ksb,
                                                 const bf16* __restrict__ vt,
                                                 const float* __restrict__ bias,
                                                 bf16* __restrict__ xo) {
  const int i = blockIdx.x;      // 13 row blocks
  const int h = blockIdx.y;      // 16 heads
  const int b = blockIdx.z;      // 256 batches
  const int lane = threadIdx.x;
  const int half = lane >> 4;
  const int nl = lane & 15;
  const size_t bh = (size_t)b * HEADS + h;

  const bf16* qbase = qs + (bh * NPADQ + (size_t)i * 16) * HD;
  const bf16* kbase = ksb + bh * NPADK * HD;
  const bf16* vbase = vt + bh * HD * NPADK;

  v16bf aq = load_a_tile(qbase, HD, lane);

  v8f s[14];
#pragma unroll
  for (int j = 0; j < 14; j++) {
    v16bf bk = load_b_tile(kbase + (size_t)j * 16 * HD, HD, lane);
    v8f z = {};
    s[j] = WMMA_BF16(aq, bk, z);
  }

  // relative-position bias (precomputed, padded with -1e30)
  const float* bbase = bias + ((size_t)h * NPADQ + (size_t)i * 16) * NPADK;
#pragma unroll
  for (int j = 0; j < 14; j++)
#pragma unroll
    for (int r = 0; r < 8; r++)
      s[j][r] += bbase[(r + 8 * half) * NPADK + j * 16 + nl];

  // softmax: row r (lanes 0-15) / row r+8 (lanes 16-31); cols across 16 lanes
  float rsum[8];
#pragma unroll
  for (int r = 0; r < 8; r++) {
    float mx = -3.0e38f;
#pragma unroll
    for (int j = 0; j < 14; j++) mx = fmaxf(mx, s[j][r]);
    for (int msk = 1; msk < 16; msk <<= 1) mx = fmaxf(mx, __shfl_xor(mx, msk, 32));
    float sm = 0.f;
#pragma unroll
    for (int j = 0; j < 14; j++) {
      float e = __expf(s[j][r] - mx);
      s[j][r] = e;
      sm += e;
    }
    for (int msk = 1; msk < 16; msk <<= 1) sm += __shfl_xor(sm, msk, 32);
    rsum[r] = sm;
  }

  // stage unnormalized P as bf16 in LDS for A-operand reload
  __shared__ __align__(16) bf16 plds[16 * NPADK];
#pragma unroll
  for (int j = 0; j < 14; j++)
#pragma unroll
    for (int r = 0; r < 8; r++)
      plds[(r + 8 * half) * NPADK + j * 16 + nl] = (bf16)s[j][r];
  __syncthreads();

  v8f o0 = {}, o1 = {};
  for (int kk = 0; kk < NPADK; kk += 32) {
    v16bf ap  = load_a_tile(plds + kk, NPADK, lane);
    v16bf bv0 = load_b_tile(vbase + kk, NPADK, lane);
    v16bf bv1 = load_b_tile(vbase + (size_t)16 * NPADK + kk, NPADK, lane);
    o0 = WMMA_BF16(ap, bv0, o0);
    o1 = WMMA_BF16(ap, bv1, o1);
  }

#pragma unroll
  for (int r = 0; r < 8; r++) {
    int tok = i * 16 + r + 8 * half;
    if (tok < NTOK) {
      float inv = 1.0f / rsum[r];
      size_t orow = ((size_t)b * NTOK + tok) * CDIM + h * HD;
      xo[orow + nl]      = (bf16)(o0[r] * inv);
      xo[orow + 16 + nl] = (bf16)(o1[r] * inv);
    }
  }
}

// ---------------------------------------------------------------------------
// Output projection: out = xo @ proj_w^T + proj_b (fp32 out).
// Same 128x128 LDS-tiled async-staged GEMM structure as qkv_gemm.
// ---------------------------------------------------------------------------
__global__ void __launch_bounds__(256) proj_gemm(const bf16* __restrict__ xo,
                                                 const bf16* __restrict__ pwb,
                                                 const float* __restrict__ pb,
                                                 float* __restrict__ out) {
  __shared__ __align__(16) bf16 ldsA[2][128 * 32];
  __shared__ __align__(16) bf16 ldsB[2][128 * 32];
  const int tid = threadIdx.x;
  const int lane = tid & 31;
  const int wid = tid >> 5;
  const int wm = wid & 3;
  const int wn = wid >> 2;
  const int m0 = blockIdx.x * 128;
  const int n0 = blockIdx.y * 128;

  v8f acc[2][4] = {};

  stage_tile(xo + (size_t)m0 * CDIM, CDIM, 0, ldsA[0], tid);
  stage_tile(pwb + (size_t)n0 * CDIM, CDIM, 0, ldsB[0], tid);
  wait_async0();
  __syncthreads();

  const int KSTEPS = CDIM / 32;
  for (int step = 0; step < KSTEPS; step++) {
    const int cur = step & 1;
    if (step + 1 < KSTEPS) {
      const int kk = (step + 1) * 32;
      stage_tile(xo + (size_t)m0 * CDIM, CDIM, kk, ldsA[cur ^ 1], tid);
      stage_tile(pwb + (size_t)n0 * CDIM, CDIM, kk, ldsB[cur ^ 1], tid);
    }
    v16bf a0 = load_a_tile(ldsA[cur] + (wm * 32 + 0) * 32, 32, lane);
    v16bf a1 = load_a_tile(ldsA[cur] + (wm * 32 + 16) * 32, 32, lane);
#pragma unroll
    for (int t = 0; t < 4; t++) {
      v16bf bt = load_b_tile(ldsB[cur] + (wn * 64 + t * 16) * 32, 32, lane);
      acc[0][t] = WMMA_BF16(a0, bt, acc[0][t]);
      acc[1][t] = WMMA_BF16(a1, bt, acc[1][t]);
    }
    wait_async0();
    __syncthreads();
  }

  const int half = lane >> 4;
  const int nl = lane & 15;
#pragma unroll
  for (int a = 0; a < 2; a++) {
#pragma unroll
    for (int t = 0; t < 4; t++) {
#pragma unroll
      for (int r = 0; r < 8; r++) {
        int m = m0 + wm * 32 + a * 16 + r + 8 * half;
        int j = n0 + wn * 64 + t * 16 + nl;
        out[(size_t)m * CDIM + j] = acc[a][t][r] + pb[j];
      }
    }
  }
}

// ---------------------------------------------------------------------------
extern "C" void kernel_launch(void* const* d_in, const int* in_sizes, int n_in,
                              void* d_out, int out_size, void* d_ws, size_t ws_size,
                              hipStream_t stream) {
  const float* x      = (const float*)d_in[0];
  const float* qkv_w  = (const float*)d_in[1];
  const float* proj_w = (const float*)d_in[2];
  const float* proj_b = (const float*)d_in[3];
  const float* table  = (const float*)d_in[4];
  const int*   relidx = (const int*)d_in[5];
  float* out = (float*)d_out;

  char* ws = (char*)d_ws;
  bf16*  xb   = (bf16*)(ws + O_XB);    // also aliased as attention output xo
  bf16*  wqb  = (bf16*)(ws + O_WQB);
  bf16*  pwb  = (bf16*)(ws + O_PWB);
  bf16*  qs   = (bf16*)(ws + O_QS);
  bf16*  ksb  = (bf16*)(ws + O_KS);
  bf16*  vt   = (bf16*)(ws + O_VT);
  float* bias = (float*)(ws + O_BIAS);

  // 1) casts to bf16
  {
    int n = MROWS * CDIM;
    cast_f32_to_bf16<<<(n + 255) / 256, 256, 0, stream>>>(x, xb, n);
    n = 3 * CDIM * CDIM;
    cast_f32_to_bf16<<<(n + 255) / 256, 256, 0, stream>>>(qkv_w, wqb, n);
    n = CDIM * CDIM;
    cast_f32_to_bf16<<<(n + 255) / 256, 256, 0, stream>>>(proj_w, pwb, n);
  }

  // 2) zero padded q/k/v^T region (qs,ksb,vt contiguous)
  {
    long n4 = (long)((QS_BYTES + KS_BYTES + VT_BYTES) / 16);
    zero_u4<<<2048, 256, 0, stream>>>((uint4*)qs, n4);
  }

  // 3) padded relative-position bias
  {
    int total = HEADS * NPADQ * NPADK;
    build_bias<<<(total + 255) / 256, 256, 0, stream>>>(table, relidx, bias);
  }

  // 4) QKV GEMM -> q (scaled), k, v^T
  {
    dim3 grid(MROWS / 128, (3 * CDIM) / 128);  // 392 x 12
    qkv_gemm<<<grid, 256, 0, stream>>>(xb, wqb, qs, ksb, vt);
  }

  // 5) fused attention -> xo (aliases xb; fully overwritten)
  {
    dim3 grid(NPADQ / 16, HEADS, BATCH);       // 13 x 16 x 256
    attn_fused<<<grid, 32, 0, stream>>>(qs, ksb, vt, bias, xb);
  }

  // 6) output projection + bias -> fp32 out
  {
    dim3 grid(MROWS / 128, CDIM / 128);        // 392 x 4
    proj_gemm<<<grid, 256, 0, stream>>>(xb, pwb, proj_b, out);
  }
}
